// AdaAttN_75316546503095
// MI455X (gfx1250) — matrix operand order repaired and checked by reading, not compiled, not run on hardware
//
#include <hip/hip_runtime.h>

// ---------------------------------------------------------------------------
// AdaAttN for MI455X (gfx1250, wave32, WMMA + TDM).
//   B=4, C=CK=512, HW=64 -> N=4096 pixels.
// Pipeline:
//   1) conv_wmma_kernel x3 : bf16 WMMA GEMMs (f/g/h 1x1 convs), outputs packed
//      in WMMA-fragment-friendly layouts.
//   2) norm_stats_kernel   : per-(b,c) mean & 1/sqrt(var+eps) of content.
//   3) adaattn_kernel      : per 16-query block; Q tile staged into LDS via
//      the Tensor Data Mover, QK^T scores kept in a 256KB LDS stripe (fits
//      320KB/WGP), triple softmax on VALU, P@V and P@V^2 via bf16 WMMA,
//      fused instance-norm epilogue.
// ---------------------------------------------------------------------------

typedef __attribute__((ext_vector_type(16))) __bf16 v16bf;
typedef __attribute__((ext_vector_type(8)))  __bf16 bf16x8;
typedef __attribute__((ext_vector_type(8)))  float  v8f;
typedef unsigned int u32x4 __attribute__((ext_vector_type(4)));
typedef int          i32x8 __attribute__((ext_vector_type(8)));
typedef int          i32x4 __attribute__((ext_vector_type(4)));

#define NBATCH 4
#define NCH    512
#define NPIX   4096
#define EPSV   1e-5f

// LDS byte offsets inside the dynamic-LDS block of adaattn_kernel.
#define SC_BYTES   (16 * NPIX * 4)            // 262144
#define QS_BYTES   (16 * NCH * 2)             // 16384

#if defined(__gfx1250__) && \
    __has_builtin(__builtin_amdgcn_tensor_load_to_lds) && \
    __has_builtin(__builtin_amdgcn_s_wait_tensorcnt)
#define USE_TDM_STAGE 1
#else
#define USE_TDM_STAGE 0
#endif

__device__ __forceinline__ __bf16 f2bf(float f) {
  unsigned u = __float_as_uint(f);
  u = (u + 0x7FFFu + ((u >> 16) & 1u)) >> 16;   // round-to-nearest-even
  return __builtin_bit_cast(__bf16, (unsigned short)u);
}
__device__ __forceinline__ float bf2f(__bf16 h) {
  unsigned short s = __builtin_bit_cast(unsigned short, h);
  return __uint_as_float(((unsigned)s) << 16);
}
__device__ __forceinline__ v16bf mk16(bf16x8 lo, bf16x8 hi) {
  v16bf r;
#pragma unroll
  for (int i = 0; i < 8; ++i) { r[i] = lo[i]; r[i + 8] = hi[i]; }
  return r;
}
__device__ __forceinline__ v8f zero8() {
  v8f z = {0.f, 0.f, 0.f, 0.f, 0.f, 0.f, 0.f, 0.f};
  return z;
}

// ---------------------------------------------------------------------------
// Kernel 1: out[o,n] = sum_c W[o,c] * X[b,c,n] + bias[o], bf16 output.
//   mode 0: out[(b*4096+n)*512 + o]                       (query-major, for Fq)
//   mode 1: out[((b*16 + o/32)*4096 + n)*32 + o%32]       (packed32, for Gp/Vp)
// Block = 128 thr (4 waves). Block tile 64(o) x 64(n); wave w owns o-rows
// [16w,16w+16) x 64 n-cols -> 4 wmma accumulators, K looped in chunks of 32.
// ---------------------------------------------------------------------------
__global__ __launch_bounds__(128)
void conv_wmma_kernel(const float* __restrict__ X, const float* __restrict__ W,
                      const float* __restrict__ bias, __bf16* __restrict__ out,
                      int mode) {
  __shared__ __bf16 Wt[64][32];   // [o][k]  bf16
  __shared__ __bf16 XtT[64][32];  // [n][k]  bf16 (transposed on fill)

  const int nBase = blockIdx.x * 64;
  const int oBase = blockIdx.y * 64;
  const int bb    = blockIdx.z;
  const int t     = threadIdx.x;
  const int lane  = t & 31;
  const int w     = t >> 5;
  const int m15   = lane & 15;
  const bool hi   = lane >= 16;

  v8f acc[4];
#pragma unroll
  for (int i = 0; i < 4; ++i) acc[i] = zero8();

  for (int kc = 0; kc < 16; ++kc) {
    __syncthreads();
    {  // W tile: 64 rows x 32 cols, 2 threads/row, 16 cols each
      int r = t >> 1, c0 = (t & 1) * 16;
      const float* src = W + (size_t)(oBase + r) * NCH + kc * 32 + c0;
#pragma unroll
      for (int i = 0; i < 16; i += 4) {
        float4 v = *(const float4*)(src + i);
        Wt[r][c0 + i + 0] = f2bf(v.x);
        Wt[r][c0 + i + 1] = f2bf(v.y);
        Wt[r][c0 + i + 2] = f2bf(v.z);
        Wt[r][c0 + i + 3] = f2bf(v.w);
      }
    }
    {  // X tile transposed: 32 k-rows x 64 n-cols, 4 threads/row, 16 n each
      int kr = t >> 2, n0 = (t & 3) * 16;
      const float* src =
          X + ((size_t)bb * NCH + kc * 32 + kr) * NPIX + nBase + n0;
      if (kc + 1 < 16)  // prefetch next K-chunk of the activation stream
        __builtin_prefetch(src + 32 * NPIX, 0, 0);
#pragma unroll
      for (int i = 0; i < 16; i += 4) {
        float4 v = *(const float4*)(src + i);
        XtT[n0 + i + 0][kr] = f2bf(v.x);
        XtT[n0 + i + 1][kr] = f2bf(v.y);
        XtT[n0 + i + 2][kr] = f2bf(v.z);
        XtT[n0 + i + 3][kr] = f2bf(v.w);
      }
    }
    __syncthreads();

    // A fragment (16-bit A layout: lanes<16 K in {0..7,16..23}, else {8..15,24..31})
    const int kbA = hi ? 8 : 0;
    const __bf16* wrow = &Wt[w * 16 + m15][0];
    bf16x8 alo = *(const bf16x8*)(wrow + kbA);
    bf16x8 ahi = *(const bf16x8*)(wrow + 16 + kbA);
    v16bf af = mk16(alo, ahi);

    const int kbB = hi ? 8 : 0;
#pragma unroll
    for (int nt = 0; nt < 4; ++nt) {
      const __bf16* xrow = &XtT[nt * 16 + m15][kbB];
      bf16x8 blo = *(const bf16x8*)(xrow);
      bf16x8 bhi = *(const bf16x8*)(xrow + 16);
      v16bf bfr = mk16(blo, bhi);
      acc[nt] = __builtin_amdgcn_wmma_f32_16x16x32_bf16(
          false, af, false, bfr, (short)0, acc[nt], false, false);
    }
  }

  // Epilogue: D element r -> o = oBase + 16w + (hi?8:0) + r (8 contiguous o).
  const int oStart = oBase + w * 16 + (hi ? 8 : 0);
#pragma unroll
  for (int nt = 0; nt < 4; ++nt) {
    int n = nBase + nt * 16 + m15;
    bf16x8 ov;
#pragma unroll
    for (int r = 0; r < 8; ++r) ov[r] = f2bf(acc[nt][r] + bias[oStart + r]);
    __bf16* dst;
    if (mode == 0)
      dst = out + ((size_t)bb * NPIX + n) * NCH + oStart;
    else
      dst = out + (((size_t)bb * 16 + (oStart >> 5)) * NPIX + n) * 32 +
            (oStart & 31);
    *(bf16x8*)dst = ov;
  }
}

// ---------------------------------------------------------------------------
// Kernel 2: per-(b,c) mean and 1/sqrt(var+eps) over 4096 pixels (ddof=1).
// ---------------------------------------------------------------------------
__global__ __launch_bounds__(256)
void norm_stats_kernel(const float* __restrict__ content,
                       float* __restrict__ cmean, float* __restrict__ cinv) {
  __shared__ float s1[256], s2[256];
  const int bc = blockIdx.x;
  const float* x = content + (size_t)bc * NPIX;
  float a = 0.f, q = 0.f;
  for (int j = threadIdx.x; j < NPIX; j += 256) {
    float v = x[j];
    a += v;
    q += v * v;
  }
  s1[threadIdx.x] = a;
  s2[threadIdx.x] = q;
  __syncthreads();
  for (int off = 128; off > 0; off >>= 1) {
    if (threadIdx.x < off) {
      s1[threadIdx.x] += s1[threadIdx.x + off];
      s2[threadIdx.x] += s2[threadIdx.x + off];
    }
    __syncthreads();
  }
  if (threadIdx.x == 0) {
    float mean = s1[0] / (float)NPIX;
    float var  = (s2[0] - (float)NPIX * mean * mean) / (float)(NPIX - 1);
    cmean[bc] = mean;
    cinv[bc]  = rsqrtf(var + EPSV);
  }
}

// ---------------------------------------------------------------------------
// Kernel 3: fused attention. One block = 16 queries of one batch, 256 thr
// (8 waves). LDS: 16x4096 f32 score stripe (256KB) + 16x512 bf16 Q (16KB).
// ---------------------------------------------------------------------------
__global__ __launch_bounds__(256)
void adaattn_kernel(const __bf16* __restrict__ Fq, const __bf16* __restrict__ Gp,
                    const __bf16* __restrict__ Vp,
                    const float* __restrict__ content,
                    const float* __restrict__ cmean,
                    const float* __restrict__ cinv,
                    const float* __restrict__ wmix, float* __restrict__ outp) {
  extern __shared__ char smem[];
  float*  sc  = (float*)smem;                       // [16][4096]
  __bf16* Qs  = (__bf16*)(smem + SC_BYTES);         // [16][512]
  float*  red = (float*)(smem + SC_BYTES + QS_BYTES);  // [512]
  float*  Md  = red + 512;
  float*  Ms  = Md + 16;
  float*  Cd  = Ms + 16;
  float*  Cs  = Cd + 16;
  float*  Zs  = Cs + 16;

  const int blk  = blockIdx.x;
  const int bb   = blk >> 8;           // 256 query-tiles per batch
  const int q0   = (blk & 255) * 16;
  const int t    = threadIdx.x;
  const int lane = t & 31;
  const int w    = t >> 5;
  const int m15  = lane & 15;
  const bool hi  = lane >= 16;
  const int rowOff = hi ? 8 : 0;

  // ---- Stage Q tile (16 x 512 bf16 = 16KB, contiguous) into LDS ----
#if USE_TDM_STAGE
  // One TDM descriptor issued by wave 0: 1-D tile of 8192 x 2-byte elements.
  // D# built per ISA 8.3/8.4. Dynamic-LDS block is assumed to start at LDS
  // offset 0, so Qs sits at byte offset SC_BYTES. This toolchain exposes the
  // 6-argument builtin (g0 x4, g1 x8, g2 x4, g3 x4, extra x8, cpol).
  if (w == 0) {
    unsigned long long ga =
        (unsigned long long)(const void*)(Fq + ((size_t)bb * NPIX + q0) * NCH);
    u32x4 g0;
    g0[0] = 1u;                                   // count=1, user descriptor
    g0[1] = (unsigned)SC_BYTES;                   // lds_addr
    g0[2] = (unsigned)(ga & 0xFFFFFFFFu);         // global_addr[31:0]
    g0[3] = (unsigned)((ga >> 32) & 0x01FFFFFFu)  // global_addr[56:32]
            | (2u << 30);                         // type=2 ("image")
    i32x8 g1;
    g1[0] = (int)(1u << 16);           // data_size=1 (2 bytes), no multicast
    g1[1] = (int)(0x2000u << 16);      // tensor_dim0 = 8192 (bits 63:48)
    g1[2] = (int)(1u << 16);           // tensor_dim0 hi=0, tensor_dim1 = 1
    g1[3] = (int)(0x2000u << 16);      // tensor_dim1 hi=0, tile_dim0 = 8192
    g1[4] = 1;                         // tile_dim1 = 1, tile_dim2 = 0
    g1[5] = 0x2000;                    // tensor_dim0_stride = 8192
    g1[6] = (int)(0x2000u << 16);      // tensor_dim1_stride = 8192
    g1[7] = 0;
    i32x4 g2 = {0, 0, 0, 0};
    i32x4 g3 = {0, 0, 0, 0};
    i32x8 g4 = {0, 0, 0, 0, 0, 0, 0, 0};
    __builtin_amdgcn_tensor_load_to_lds(g0, g1, g2, g3, g4, 0);
    __builtin_amdgcn_s_wait_tensorcnt(0);
  }
#else
  for (int i = t; i < 16 * NCH / 8; i += 256) {
    int r = i >> 6, c0 = (i & 63) * 8;
    *(bf16x8*)&Qs[r * NCH + c0] =
        *(const bf16x8*)(Fq + ((size_t)bb * NPIX + q0 + r) * NCH + c0);
  }
#endif
  __syncthreads();

  // ---- QK^T: 16 x 4096 scores -> LDS (512 wmma per wave) ----
  for (int n0 = w * 16; n0 < NPIX; n0 += 128) {
    if (n0 + 128 < NPIX)  // prefetch next stripe of the K stream into GL2
      __builtin_prefetch(
          Gp + (((size_t)bb * 16) * NPIX + n0 + 128 + m15) * 32, 0, 0);
    v8f acc = zero8();
#pragma unroll
    for (int kc = 0; kc < 16; ++kc) {
      const int kbA = hi ? 8 : 0;
      const __bf16* qrow = Qs + m15 * NCH + kc * 32;
      bf16x8 alo = *(const bf16x8*)(qrow + kbA);
      bf16x8 ahi = *(const bf16x8*)(qrow + 16 + kbA);
      v16bf af = mk16(alo, ahi);
      const int kbB = hi ? 8 : 0;
      const __bf16* gsrc =
          Gp + (((size_t)bb * 16 + kc) * NPIX + n0 + m15) * 32 + kbB;
      bf16x8 blo = *(const bf16x8*)(gsrc);
      bf16x8 bhi = *(const bf16x8*)(gsrc + 16);
      v16bf bfr = mk16(blo, bhi);
      acc = __builtin_amdgcn_wmma_f32_16x16x32_bf16(false, af, false, bfr,
                                                    (short)0, acc, false, false);
    }
#pragma unroll
    for (int r = 0; r < 8; ++r)
      sc[(size_t)(rowOff + r) * NPIX + n0 + m15] = acc[r];
  }
  __syncthreads();

  // ---- softmax stats: 16 threads per query row ----
  {
    const int row = t >> 4, sub = t & 15;
    float* srow = sc + (size_t)row * NPIX;

    float mx = -1e30f;
    for (int j = sub; j < NPIX; j += 16) mx = fmaxf(mx, srow[j]);
    red[row * 16 + sub] = mx;
    __syncthreads();
    if (sub == 0) {
      float m = red[row * 16];
      for (int i = 1; i < 16; ++i) m = fmaxf(m, red[row * 16 + i]);
      Md[row] = m;
      Ms[row] = fmaxf(m, 0.f);  // max of relu(a)
    }
    __syncthreads();

    const float md = Md[row], ms = Ms[row];
    float sd = 0.f, ss = 0.f;
    for (int j = sub; j < NPIX; j += 16) {
      float a = srow[j];
      sd += __expf(a - md);
      ss += __expf(fmaxf(a, 0.f) - ms);
    }
    red[row * 16 + sub] = sd;
    red[256 + row * 16 + sub] = ss;
    __syncthreads();
    if (sub == 0) {
      float d = 0.f, s = 0.f;
      for (int i = 0; i < 16; ++i) {
        d += red[row * 16 + i];
        s += red[256 + row * 16 + i];
      }
      float e0 = __expf(wmix[0]), e1 = __expf(wmix[1]);
      float inv = 1.f / (e0 + e1);
      Cd[row] = (e0 * inv) / d;   // w0 / sumexp_dense
      Cs[row] = (e1 * inv) / s;   // w1 / sumexp_sparse
    }
    __syncthreads();

    // z = w0*softmax_dense + w1*softmax_sparse  in [0,1]; outer softmax
    // needs no max-shift: overwrite scores with p = exp(z), reduce Z.
    const float cd = Cd[row], cs = Cs[row];
    float zsum = 0.f;
    for (int j = sub; j < NPIX; j += 16) {
      float a = srow[j];
      float z = cd * __expf(a - md) + cs * __expf(fmaxf(a, 0.f) - ms);
      float p = __expf(z);
      srow[j] = p;
      zsum += p;
    }
    red[row * 16 + sub] = zsum;
    __syncthreads();
    if (sub == 0) {
      float zt = 0.f;
      for (int i = 0; i < 16; ++i) zt += red[row * 16 + i];
      Zs[row] = zt;
    }
    __syncthreads();
  }

  // ---- P@V and P@V^2 : wave w owns channels [64w, 64w+64) (1024 wmma/wave) ----
  v8f macc[4], sacc[4];
#pragma unroll
  for (int i = 0; i < 4; ++i) { macc[i] = zero8(); sacc[i] = zero8(); }
  const int c0 = w * 64;

#pragma unroll 2
  for (int pc = 0; pc < NPIX / 32; ++pc) {
    if (pc + 1 < NPIX / 32)  // prefetch next V chunk into GL2
      __builtin_prefetch(
          Vp + (((size_t)bb * 128 + pc + 1) * NCH + c0 + m15) * 32, 0, 0);
    const int kbA = hi ? 8 : 0;
    const float* prow = sc + (size_t)m15 * NPIX + pc * 32;
    v16bf pf;
#pragma unroll
    for (int e = 0; e < 8; ++e) pf[e] = f2bf(prow[kbA + e]);
#pragma unroll
    for (int e = 0; e < 8; ++e) pf[8 + e] = f2bf(prow[16 + kbA + e]);

    const int kbB = hi ? 8 : 0;
#pragma unroll
    for (int ct = 0; ct < 4; ++ct) {
      int ch = c0 + ct * 16 + m15;
      const __bf16* vsrc =
          Vp + (((size_t)bb * 128 + pc) * NCH + ch) * 32 + kbB;
      bf16x8 vlo = *(const bf16x8*)(vsrc);
      bf16x8 vhi = *(const bf16x8*)(vsrc + 16);
      v16bf vf = mk16(vlo, vhi);
      macc[ct] = __builtin_amdgcn_wmma_f32_16x16x32_bf16(
          false, pf, false, vf, (short)0, macc[ct], false, false);
      v16bf v2;
#pragma unroll
      for (int e = 0; e < 16; ++e) {
        float f = bf2f(vf[e]);
        v2[e] = f2bf(f * f);
      }
      sacc[ct] = __builtin_amdgcn_wmma_f32_16x16x32_bf16(
          false, pf, false, v2, (short)0, sacc[ct], false, false);
    }
  }

  // ---- epilogue: std * (content - cmean)*cinv + mean ----
#pragma unroll
  for (int ct = 0; ct < 4; ++ct) {
    int ch = c0 + ct * 16 + m15;
    size_t base = ((size_t)bb * NCH + ch) * NPIX + q0;
    float cm = cmean[bb * NCH + ch];
    float ci = cinv[bb * NCH + ch];
#pragma unroll
    for (int r = 0; r < 8; ++r) {
      int qr = rowOff + r;
      float invZ = 1.f / Zs[qr];
      float mean = macc[ct][r] * invZ;
      float sec  = sacc[ct][r] * invZ;
      float stdv = sqrtf(fmaxf(sec - mean * mean, 0.f));
      float x = content[base + qr];
      outp[base + qr] = stdv * (x - cm) * ci + mean;
    }
  }
}

// ---------------------------------------------------------------------------
extern "C" void kernel_launch(void* const* d_in, const int* in_sizes, int n_in,
                              void* d_out, int out_size, void* d_ws,
                              size_t ws_size, hipStream_t stream) {
  const float* content = (const float*)d_in[0];
  const float* style   = (const float*)d_in[1];
  const float* ckey    = (const float*)d_in[2];
  const float* skey    = (const float*)d_in[3];
  const float* f_w = (const float*)d_in[4];
  const float* f_b = (const float*)d_in[5];
  const float* g_w = (const float*)d_in[6];
  const float* g_b = (const float*)d_in[7];
  const float* h_w = (const float*)d_in[8];
  const float* h_b = (const float*)d_in[9];
  const float* wmix = (const float*)d_in[10];

  // Workspace: 3 x 16.78MB bf16 tensors + 16KB stats (~50.4MB total).
  char* ws = (char*)d_ws;
  const size_t TSZ = (size_t)NBATCH * NPIX * NCH * sizeof(unsigned short);
  __bf16* Fq = (__bf16*)ws;                 // [b][n][512]      (A layout)
  __bf16* Gp = (__bf16*)(ws + TSZ);         // [b][kc][n][32]   (B layout)
  __bf16* Vp = (__bf16*)(ws + 2 * TSZ);     // [b][pc][ch][32]  (B layout)
  float* cmean = (float*)(ws + 3 * TSZ);    // [b*512]
  float* cinv  = cmean + NBATCH * NCH;      // [b*512]
  float* outp  = (float*)d_out;

  dim3 cgrid(NPIX / 64, NCH / 64, NBATCH);
  conv_wmma_kernel<<<cgrid, 128, 0, stream>>>(ckey, f_w, f_b, Fq, 0);
  conv_wmma_kernel<<<cgrid, 128, 0, stream>>>(skey, g_w, g_b, Gp, 1);
  conv_wmma_kernel<<<cgrid, 128, 0, stream>>>(style, h_w, h_b, Vp, 1);

  norm_stats_kernel<<<NBATCH * NCH, 256, 0, stream>>>(content, cmean, cinv);

  // LDS: 256KB scores + 16KB Q + reduction/stat scratch (< 320KB/WGP).
  size_t smem = (size_t)SC_BYTES + (size_t)QS_BYTES + (512 + 5 * 16) * 4;
  adaattn_kernel<<<NBATCH * (NPIX / 16), 256, smem, stream>>>(
      Fq, Gp, Vp, content, cmean, cinv, wmix, outp);
}
